// FrequencyFilter_3049426780642
// MI455X (gfx1250) — matrix operand order repaired and checked by reading, not compiled, not run on hardware
//
#include <hip/hip_runtime.h>
#include <hip/hip_bf16.h>
#include <cstdint>
#include <cstddef>

// ---------------- problem constants ----------------
#define T_LEN    100000          // time samples
#define C_SEL    256             // selected channels
#define C_IN     512             // input channels
#define KW       12000           // gaussian taps
#define PW       5999            // SAME left padding = (KW-1)/2
#define PADT     6144            // zero pad (time) on each side of staged input
#define TPAD     (T_LEN + 2*PADT)   // 112288 (multiple of 8 -> 16B aligned rows)
#define WREP_PAD 160             // lead pad (halves) in each phase-shifted w copy
#define WREP_LEN 12320           // halves per phase row (multiple of 16)
#define JSTART   (-6016)         // band start rel. to workgroup t0 (multiple of 32)
#define NIT      380             // k-loop iterations: covers [-6016, 6143] rel. window
#define T_WG     128             // output times per workgroup
#define WSCALE   1024.0f         // f16 weight scaling (keeps taps in normal range)

typedef __attribute__((ext_vector_type(16))) _Float16 v16h;
typedef __attribute__((ext_vector_type(8)))  _Float16 v8h;
typedef __attribute__((ext_vector_type(8)))  float    v8f;
typedef __attribute__((ext_vector_type(4)))  float    v4f;

// =====================================================================
// Core kernel: banded-Toeplitz correlation as WMMA GEMM.
//   xt   : f16 staged input, channel-major [C_SEL][TPAD], valid t at j=PADT+t
//   wrep : f16 weights*WSCALE, 16 phase-shifted copies [16][WREP_LEN]
//   out  : f32 [T_LEN][C_SEL]
// Workgroup: 256 threads = 8 waves; tile = 64 channels x 128 times.
// Wave w: channel group cg=w&3 (16 ch), time half th=w>>2 (4 groups of 16 t).
// Per iteration: A (16ch x 32t data) prefetched via async-to-LDS double
// buffer; 4 Toeplitz B tiles (32t x 16t) from aligned global loads; 4 WMMAs.
// =====================================================================
__global__ __launch_bounds__(256) void conv_wmma_kernel(
    const _Float16* __restrict__ xt,
    const _Float16* __restrict__ wrep,
    float* __restrict__ out)
{
    __shared__ __align__(32) char asmem[16384];   // 8 waves * 2 bufs * 32 lanes * 32B

    const int tid  = threadIdx.x;
    const int lane = tid & 31;
    const int wv   = tid >> 5;          // wave id 0..7
    const int cg   = wv & 3;            // channel group in workgroup
    const int th   = wv >> 2;           // time half (0/1)
    const int N    = lane & 15;
    const int hi   = (lane >> 4) & 1;

    const int t0_wg = blockIdx.x * T_WG;
    const int ch0   = blockIdx.y * 64 + cg * 16;

    // ---- A (data) per-lane global source: contiguous k = 16*hi + s order, same as B.
    // lane<16 holds K=0..15 (contiguous) of channel row N;
    // lane>=16 holds K=16..31  ->  +16 halves base offset.
    const char* gp = (const char*)xt +
        2u * ((size_t)(ch0 + N) * TPAD + (size_t)(PADT + t0_wg + JSTART) + (hi ? 16 : 0));

    // ---- LDS double-buffered staging slot for this lane (32B per buffer)
    char* lw = asmem + wv * 2048 + lane * 32;
    const uint32_t lds_lo = (uint32_t)(uintptr_t)lw;   // low 32 bits = LDS offset

    // ---- B (Toeplitz weights): per-lane 16 consecutive taps starting at
    // s = PW + jb - t0 + Kbase - N.  s mod 16 depends only on the lane, so a
    // phase-shifted copy of w gives 32B-aligned v16h loads.
    const int ph  = (15 - N) & 15;
    const int s00 = PW + JSTART - 64 * th + (hi ? 16 : 0) - N;   // gi=0, it=0
    const char* bp = (const char*)wrep +
        2u * ((size_t)ph * WREP_LEN + (size_t)(WREP_PAD + s00 - ph));

    v8f acc[4] = {};

    // prologue: prefetch iteration 0 into buffer 0 (ASYNCcnt += 2)
    {
        const uint64_t g0 = (uint64_t)(uintptr_t)gp;
        const uint64_t g1 = g0 + 16;
        asm volatile("global_load_async_to_lds_b128 %0, %1, off"
                     :: "v"(lds_lo), "v"(g0) : "memory");
        asm volatile("global_load_async_to_lds_b128 %0, %1, off"
                     :: "v"(lds_lo + 16u), "v"(g1) : "memory");
    }

    for (int it = 0; it < NIT; ++it) {
        const uint32_t cur = (uint32_t)(it & 1) * 1024u;
        const uint32_t nxt = (uint32_t)((it + 1) & 1) * 1024u;

        // prefetch next iteration (last one overruns into in-bounds tail)
        {
            const uint64_t g0 = (uint64_t)(uintptr_t)(gp + (size_t)(it + 1) * 64);
            const uint64_t g1 = g0 + 16;
            asm volatile("global_load_async_to_lds_b128 %0, %1, off"
                         :: "v"(lds_lo + nxt), "v"(g0) : "memory");
            asm volatile("global_load_async_to_lds_b128 %0, %1, off"
                         :: "v"(lds_lo + nxt + 16u), "v"(g1) : "memory");
        }
        // wait until current buffer's two async copies have landed in LDS
        asm volatile("s_wait_asynccnt 0x2" ::: "memory");

        const v16h a = *(const v16h*)(lw + cur);          // ds_load_b128 x2

        const char* bit = bp + (size_t)it * 64;
        #pragma unroll
        for (int gi = 0; gi < 4; ++gi) {
            const v16h b = *(const v16h*)(bit - gi * 32); // 32B-aligned global
            acc[gi] = __builtin_amdgcn_wmma_f32_16x16x32_f16(
                false, a, false, b, (short)0, acc[gi], false, false);
        }
    }

    // ---- store D: VGPR r, lane<16 -> (M=r, n=lane); lane>=16 -> (M=8+r).
    // M = channel offset, n = time offset  => per-lane 8 contiguous floats.
    const float sc = 1.0f / WSCALE;
    const int cbase = ch0 + hi * 8;
    #pragma unroll
    for (int gi = 0; gi < 4; ++gi) {
        const int t = t0_wg + th * 64 + gi * 16 + N;
        if (t < T_LEN) {
            float* op = out + (size_t)t * C_SEL + cbase;
            const v8f a8 = acc[gi];
            v4f o0 = { a8[0]*sc, a8[1]*sc, a8[2]*sc, a8[3]*sc };
            v4f o1 = { a8[4]*sc, a8[5]*sc, a8[6]*sc, a8[7]*sc };
            *(v4f*)op       = o0;
            *(v4f*)(op + 4) = o1;
        }
    }
}

// =====================================================================
// Stage input: channel gather (+ optional high-pass filt) + f32->f16 +
// transpose to channel-major padded layout, via LDS 32x32 tiles.
// =====================================================================
__global__ __launch_bounds__(256) void stage_input_kernel(
    const float* __restrict__ firings,   // [T][512]
    const int*   __restrict__ sel_idx,   // [256]
    const float* __restrict__ low,       // [T][256] (pass 2) or nullptr
    const float* __restrict__ meanp,     // scalar (pass 2) or nullptr
    _Float16* __restrict__ xt,           // [256][TPAD]
    int with_filt)
{
    __shared__ float tile[32][33];
    const int tx = threadIdx.x;          // 0..31
    const int ty = threadIdx.y;          // 0..7
    const int t0 = blockIdx.x * 32;      // T_LEN divisible by 32
    const int c0 = blockIdx.y * 32;
    const float mean = with_filt ? meanp[0] : 0.0f;

    #pragma unroll
    for (int i = ty; i < 32; i += 8) {
        const int t = t0 + i;
        const int c = c0 + tx;
        float v = firings[(size_t)t * C_IN + sel_idx[c]];
        if (with_filt) v = v - low[(size_t)t * C_SEL + c] + mean;
        tile[i][tx] = v;
    }
    __syncthreads();
    #pragma unroll
    for (int i = ty; i < 32; i += 8) {
        const int c = c0 + i;
        const int t = t0 + tx;
        xt[(size_t)c * TPAD + (size_t)(PADT + t)] = (_Float16)tile[tx][i];
    }
}

// Zero the time pads of the staged buffer (+ tail slack) each call.
__global__ void zero_pads_kernel(_Float16* __restrict__ xt)
{
    const size_t npad  = (size_t)C_SEL * 2 * PADT;
    const size_t slack = 2048;  // halves (4096 B tail slack)
    for (size_t i = (size_t)blockIdx.x * blockDim.x + threadIdx.x;
         i < npad + slack; i += (size_t)gridDim.x * blockDim.x) {
        if (i < npad) {
            const size_t c = i / (2 * PADT);
            const size_t k = i % (2 * PADT);
            const size_t j = (k < PADT) ? k : ((size_t)PADT + T_LEN + (k - PADT));
            xt[c * TPAD + j] = (_Float16)0.0f;
        } else {
            xt[(size_t)C_SEL * TPAD + (i - npad)] = (_Float16)0.0f;
        }
    }
}

// Build 16 phase-shifted, zero-padded, WSCALE-scaled f16 copies of w.
__global__ void build_wrep_kernel(const float* __restrict__ w,
                                  _Float16* __restrict__ wrep)
{
    const int total = 16 * WREP_LEN;
    for (int i = blockIdx.x * blockDim.x + threadIdx.x; i < total;
         i += gridDim.x * blockDim.x) {
        const int ph = i / WREP_LEN;
        const int j  = i % WREP_LEN - WREP_PAD;
        const int wi = j + ph;
        const float v = (wi >= 0 && wi < KW) ? w[wi] * WSCALE : 0.0f;
        wrep[i] = (_Float16)v;
    }
}

// Deterministic two-stage mean of `low` (f64 accumulation).
__global__ __launch_bounds__(256) void reduce_partial_kernel(
    const float* __restrict__ in, double* __restrict__ part)
{
    __shared__ double sd[256];
    const size_t n = (size_t)T_LEN * C_SEL;
    double s = 0.0;
    for (size_t i = (size_t)blockIdx.x * blockDim.x + threadIdx.x; i < n;
         i += (size_t)gridDim.x * blockDim.x)
        s += (double)in[i];
    sd[threadIdx.x] = s;
    __syncthreads();
    for (int o = 128; o > 0; o >>= 1) {
        if ((int)threadIdx.x < o) sd[threadIdx.x] += sd[threadIdx.x + o];
        __syncthreads();
    }
    if (threadIdx.x == 0) part[blockIdx.x] = sd[0];
}

__global__ __launch_bounds__(256) void finalize_mean_kernel(
    const double* __restrict__ part, int np, float* __restrict__ meanp)
{
    __shared__ double sd[256];
    double s = 0.0;
    for (int i = threadIdx.x; i < np; i += 256) s += part[i];
    sd[threadIdx.x] = s;
    __syncthreads();
    for (int o = 128; o > 0; o >>= 1) {
        if ((int)threadIdx.x < o) sd[threadIdx.x] += sd[threadIdx.x + o];
        __syncthreads();
    }
    if (threadIdx.x == 0)
        meanp[0] = (float)(sd[0] / (double)((size_t)T_LEN * C_SEL));
}

// =====================================================================
extern "C" void kernel_launch(void* const* d_in, const int* in_sizes, int n_in,
                              void* d_out, int out_size, void* d_ws, size_t ws_size,
                              hipStream_t stream)
{
    const float* firings = (const float*)d_in[0];
    const float* glow    = (const float*)d_in[1];
    const float* ghigh   = (const float*)d_in[2];
    const int*   sel     = (const int*)d_in[3];
    float* out = (float*)d_out;

    // ---- workspace layout (all 256B-aligned) ----
    char* ws = (char*)d_ws;
    size_t off = 0;
    _Float16* xt = (_Float16*)(ws + off);
    off += (size_t)C_SEL * TPAD * 2 + 4096;          // staged input + tail slack
    off = (off + 255) & ~(size_t)255;
    _Float16* wrep_lo = (_Float16*)(ws + off);
    off += (size_t)16 * WREP_LEN * 2;
    off = (off + 255) & ~(size_t)255;
    _Float16* wrep_hi = (_Float16*)(ws + off);
    off += (size_t)16 * WREP_LEN * 2;
    off = (off + 255) & ~(size_t)255;
    double* part = (double*)(ws + off);
    off += 1024 * sizeof(double);
    off = (off + 255) & ~(size_t)255;
    float* meanp = (float*)(ws + off);

    const dim3 convGrid((T_LEN + T_WG - 1) / T_WG, C_SEL / 64);
    const dim3 stageGrid(T_LEN / 32, C_SEL / 32);
    const dim3 stageBlk(32, 8);

    // 0) zero the time pads + slack of the staged buffer
    zero_pads_kernel<<<dim3(2048), dim3(256), 0, stream>>>(xt);
    // 1) build phase-shifted f16 weight tables
    build_wrep_kernel<<<dim3(256), dim3(256), 0, stream>>>(glow,  wrep_lo);
    build_wrep_kernel<<<dim3(256), dim3(256), 0, stream>>>(ghigh, wrep_hi);
    // 2) stage sel(firings) as f16 channel-major
    stage_input_kernel<<<stageGrid, stageBlk, 0, stream>>>(
        firings, sel, nullptr, nullptr, xt, 0);
    // 3) pass 1: low = sel (x) gauss_low   (d_out used as scratch for low)
    conv_wmma_kernel<<<convGrid, dim3(256), 0, stream>>>(xt, wrep_lo, out);
    // 4) mean(low), deterministic
    reduce_partial_kernel<<<dim3(1024), dim3(256), 0, stream>>>(out, part);
    finalize_mean_kernel<<<dim3(1), dim3(256), 0, stream>>>(part, 1024, meanp);
    // 5) stage filt = sel - low + mean
    stage_input_kernel<<<stageGrid, stageBlk, 0, stream>>>(
        firings, sel, out, meanp, xt, 1);
    // 6) pass 2: out = filt (x) gauss_high
    conv_wmma_kernel<<<convGrid, dim3(256), 0, stream>>>(xt, wrep_hi, out);
}
